// MoELayer_90245852823703
// MI455X (gfx1250) — compile-verified, matrix operand order checked by reading, not run on hardware
//
#include <hip/hip_runtime.h>
#include <hip/hip_bf16.h>
#include <cstdint>

// ---- problem constants (match reference) ----
#define N_TOK 4096     // B*T
#define DIM   768      // D
#define HID   1536     // H
#define NEXP  8        // E
#define NGRP  9        // 1 shared + 8 routed
#define CSTR  16       // combine row stride (padded)

// ---- GEMM tile config ----
#define BM 128
#define BN 64
#define BK 32
#define NWAVE 8
#define TPB (NWAVE * 32)

typedef __bf16 bf16_t;
typedef __attribute__((ext_vector_type(16))) __bf16 v16bf;
typedef __attribute__((ext_vector_type(8)))  float  v8f;

// CDNA5 async global->LDS path (guarded; falls back to register staging)
#if defined(__gfx1250__) && __has_builtin(__builtin_amdgcn_global_load_async_to_lds_b128)
#define USE_ASYNC 1
#define AS1 __attribute__((address_space(1)))
#define AS3 __attribute__((address_space(3)))
typedef int async_v4i __attribute__((vector_size(16)));  // matches builtin's pointee type
#endif

__device__ __forceinline__ bf16_t f2bf(float f) {
  union { float f; uint32_t u; } v; v.f = f;
  uint32_t r = (v.u + 0x7FFFu + ((v.u >> 16) & 1u)) >> 16;  // RNE
  union { uint16_t s; bf16_t b; } o; o.s = (uint16_t)r;
  return o.b;
}

union BFrag { v16bf v; uint4 q[2]; };

// Load a 16x32 bf16 fragment from an LDS tile stored row-major [rows][BK].
// CDNA5 16-bit A/B layout: lanes 0-15 hold K in {0..7, 16..23},
// lanes 16-31 hold K in {8..15, 24..31} for the same row (lane & 15).
__device__ __forceinline__ v16bf load_frag(const bf16_t* base, int row, int lane) {
  const int k0 = (lane & 16) ? 8 : 0;
  BFrag f;
  const bf16_t* p = base + row * BK + k0;
  f.q[0] = *(const uint4*)(p);        // K = k0 .. k0+7
  f.q[1] = *(const uint4*)(p + 16);   // K = k0+16 .. k0+23
  return f.v;
}

__device__ __forceinline__ v8f wmma_bf16(v16bf a, v16bf b, v8f c) {
  return __builtin_amdgcn_wmma_f32_16x16x32_bf16(false, a, false, b,
                                                 (short)0, c, false, false);
}

// Stage a BM x BK bf16 tile (no transpose): 512 x 16B chunks, 2 per thread.
__device__ __forceinline__ void stage_a_tile(bf16_t* Ls, const bf16_t* G,
                                             long row0, long ld, int k0, int tid) {
#ifdef USE_ASYNC
#pragma unroll
  for (int i = 0; i < 2; ++i) {
    int u = tid + i * TPB;
    int r = u >> 2, c = (u & 3) * 8;
    const bf16_t* g = G + (row0 + r) * ld + k0 + c;
    __builtin_amdgcn_global_load_async_to_lds_b128(
        (AS1 async_v4i*)(uintptr_t)g,
        (AS3 async_v4i*)(uint32_t)(uintptr_t)(Ls + r * BK + c), 0, 0);
  }
#else
  const int u0 = tid, u1 = tid + TPB;
  const int r0 = u0 >> 2, c0 = (u0 & 3) * 8;
  const int r1 = u1 >> 2, c1 = (u1 & 3) * 8;
  uint4 t0 = *(const uint4*)(G + (row0 + r0) * ld + k0 + c0);
  uint4 t1 = *(const uint4*)(G + (row0 + r1) * ld + k0 + c1);
  *(uint4*)(Ls + r0 * BK + c0) = t0;
  *(uint4*)(Ls + r1 * BK + c1) = t1;
#endif
}

// Stage one 4x4 block of a BK x BN weight tile, transposed into LDS [n][k].
// blk in [0,128): kb = 4*(blk>>4), nb = 4*(blk&15).
__device__ __forceinline__ void stage_b_tile_tr(bf16_t* Ls, const bf16_t* G,
                                                int k0, long ld, int colbase, int blk) {
  const int kb = (blk >> 4) * 4, nb = (blk & 15) * 4;
  uint2 r0 = *(const uint2*)(G + (long)(k0 + kb + 0) * ld + colbase + nb);
  uint2 r1 = *(const uint2*)(G + (long)(k0 + kb + 1) * ld + colbase + nb);
  uint2 r2 = *(const uint2*)(G + (long)(k0 + kb + 2) * ld + colbase + nb);
  uint2 r3 = *(const uint2*)(G + (long)(k0 + kb + 3) * ld + colbase + nb);
  const bf16_t* e0 = (const bf16_t*)&r0;
  const bf16_t* e1 = (const bf16_t*)&r1;
  const bf16_t* e2 = (const bf16_t*)&r2;
  const bf16_t* e3 = (const bf16_t*)&r3;
#pragma unroll
  for (int j = 0; j < 4; ++j) {
    bf16_t t[4] = { e0[j], e1[j], e2[j], e3[j] };
    *(uint2*)(Ls + (nb + j) * BK + kb) = *(const uint2*)t;
  }
}

__device__ __forceinline__ void wait_stage() {
#ifdef USE_ASYNC
#if __has_builtin(__builtin_amdgcn_s_wait_asynccnt)
  __builtin_amdgcn_s_wait_asynccnt(0);
#else
  asm volatile("s_wait_asynccnt 0x0" ::: "memory");
#endif
#endif
  __syncthreads();
}

// ---------------- fp32 -> bf16 conversion ----------------
__global__ __launch_bounds__(256) void MoE_cvt(const float* __restrict__ in,
                                               bf16_t* __restrict__ out, long n) {
  long stride = (long)gridDim.x * blockDim.x;
  for (long i = (long)blockIdx.x * blockDim.x + threadIdx.x; i < n; i += stride)
    out[i] = f2bf(in[i]);
}

// ---------------- router: softmax + top-2 renorm ----------------
__global__ __launch_bounds__(256) void MoE_router(const float* __restrict__ x,
                                                  const float* __restrict__ rw,
                                                  float* __restrict__ probs,
                                                  float* __restrict__ comb) {
  int t = blockIdx.x * blockDim.x + threadIdx.x;
  if (t >= N_TOK) return;
  float acc[NEXP];
#pragma unroll
  for (int e = 0; e < NEXP; ++e) acc[e] = 0.f;
  const float* xr = x + (long)t * DIM;
  for (int i = 0; i < DIM; ++i) {
    float xv = xr[i];
#pragma unroll
    for (int e = 0; e < NEXP; ++e) acc[e] += xv * rw[e * DIM + i];
  }
  float m = acc[0];
#pragma unroll
  for (int e = 1; e < NEXP; ++e) m = fmaxf(m, acc[e]);
  float s = 0.f;
#pragma unroll
  for (int e = 0; e < NEXP; ++e) { acc[e] = __expf(acc[e] - m); s += acc[e]; }
  float inv = 1.f / s;
#pragma unroll
  for (int e = 0; e < NEXP; ++e) {
    acc[e] *= inv;
    probs[(long)t * NEXP + e] = acc[e];
  }
  int i1 = 0; float v1 = acc[0];
#pragma unroll
  for (int e = 1; e < NEXP; ++e) if (acc[e] > v1) { v1 = acc[e]; i1 = e; }
  int i2 = -1; float v2 = -1.f;
#pragma unroll
  for (int e = 0; e < NEXP; ++e)
    if (e != i1 && acc[e] > v2) { v2 = acc[e]; i2 = e; }
  float rs = 1.f / (v1 + v2);
  float* cr = comb + (long)t * CSTR;
  cr[0] = 1.0f;  // shared group weight
#pragma unroll
  for (int e = 0; e < NEXP; ++e)
    cr[1 + e] = (e == i1) ? v1 * rs : ((e == i2) ? v2 * rs : 0.f);
}

// ---------------- FFN stage 1: Y = silu(X*W1) .* (X*W2) ----------------
__global__ __launch_bounds__(TPB) void MoE_ffn1(const bf16_t* __restrict__ X,   // [N, DIM]
                                                const bf16_t* __restrict__ W1,  // [DIM, HID]
                                                const bf16_t* __restrict__ W2,  // [DIM, HID]
                                                bf16_t* __restrict__ Y) {       // [N, HID]
  __shared__ __align__(16) bf16_t Xs[2][BM * BK];
  __shared__ __align__(16) bf16_t W1s[2][BN * BK];  // transposed [n][k]
  __shared__ __align__(16) bf16_t W2s[2][BN * BK];

  const int tid = threadIdx.x;
  const int lane = tid & 31;
  const int wave = tid >> 5;               // 0..7 -> 16-row strip
  const int rowbase = blockIdx.x * BM;
  const int colbase = blockIdx.y * BN;
  const int NK = DIM / BK;

  v8f acc1[4], acc2[4];
#pragma unroll
  for (int j = 0; j < 4; ++j) { acc1[j] = (v8f)(0.f); acc2[j] = (v8f)(0.f); }

  // prologue: stage tile 0
  stage_a_tile(Xs[0], X, rowbase, DIM, 0, tid);
  if (tid < 128) stage_b_tile_tr(W1s[0], W1, 0, HID, colbase, tid);
  else           stage_b_tile_tr(W2s[0], W2, 0, HID, colbase, tid - 128);
  wait_stage();

  for (int kt = 0; kt < NK; ++kt) {
    const int cur = kt & 1, nxt = cur ^ 1;
    // issue async loads + B-transpose for tile kt+1 while computing tile kt
    if (kt + 1 < NK) {
      const int k1 = (kt + 1) * BK;
      stage_a_tile(Xs[nxt], X, rowbase, DIM, k1, tid);
      if (tid < 128) stage_b_tile_tr(W1s[nxt], W1, k1, HID, colbase, tid);
      else           stage_b_tile_tr(W2s[nxt], W2, k1, HID, colbase, tid - 128);
    }

    v16bf a = load_frag(Xs[cur], wave * 16 + (lane & 15), lane);
    v16bf b1[4], b2[4];
#pragma unroll
    for (int j = 0; j < 4; ++j) {
      b1[j] = load_frag(W1s[cur], j * 16 + (lane & 15), lane);
      b2[j] = load_frag(W2s[cur], j * 16 + (lane & 15), lane);
    }
#pragma unroll
    for (int j = 0; j < 4; ++j) acc1[j] = wmma_bf16(a, b1[j], acc1[j]);
#pragma unroll
    for (int j = 0; j < 4; ++j) acc2[j] = wmma_bf16(a, b2[j], acc2[j]);

    wait_stage();  // next tile landed + all waves done reading cur
  }

  // epilogue: SwiGLU, write bf16
  const int mbase = rowbase + wave * 16 + ((lane & 16) ? 8 : 0);
  const int n = lane & 15;
#pragma unroll
  for (int j = 0; j < 4; ++j) {
#pragma unroll
    for (int r = 0; r < 8; ++r) {
      float aa = acc1[j][r];
      float bb = acc2[j][r];
      float y = (aa / (1.f + __expf(-aa))) * bb;
      Y[(long)(mbase + r) * HID + colbase + j * 16 + n] = f2bf(y);
    }
  }
}

// ---------------- FFN stage 2: out (+)= rowscale(Y * P, comb[:,g]) ----------------
__global__ __launch_bounds__(TPB) void MoE_ffn2(const bf16_t* __restrict__ Y,   // [N, HID]
                                                const bf16_t* __restrict__ P,   // [HID, DIM]
                                                const float* __restrict__ comb, // [N, CSTR]
                                                float* __restrict__ out,        // [N, DIM]
                                                int g) {
  __shared__ __align__(16) bf16_t Ys[2][BM * BK];
  __shared__ __align__(16) bf16_t Ps[2][BN * BK];  // transposed [n][k]

  const int tid = threadIdx.x;
  const int lane = tid & 31;
  const int wave = tid >> 5;
  const int rowbase = blockIdx.x * BM;
  const int colbase = blockIdx.y * BN;
  const int NK = HID / BK;

  v8f acc[4];
#pragma unroll
  for (int j = 0; j < 4; ++j) acc[j] = (v8f)(0.f);

  stage_a_tile(Ys[0], Y, rowbase, HID, 0, tid);
  if (tid < 128) stage_b_tile_tr(Ps[0], P, 0, DIM, colbase, tid);
  wait_stage();

  for (int kt = 0; kt < NK; ++kt) {
    const int cur = kt & 1, nxt = cur ^ 1;
    if (kt + 1 < NK) {
      const int k1 = (kt + 1) * BK;
      stage_a_tile(Ys[nxt], Y, rowbase, HID, k1, tid);
      if (tid < 128) stage_b_tile_tr(Ps[nxt], P, k1, DIM, colbase, tid);
    }

    v16bf a = load_frag(Ys[cur], wave * 16 + (lane & 15), lane);
    v16bf b[4];
#pragma unroll
    for (int j = 0; j < 4; ++j) b[j] = load_frag(Ps[cur], j * 16 + (lane & 15), lane);
#pragma unroll
    for (int j = 0; j < 4; ++j) acc[j] = wmma_bf16(a, b[j], acc[j]);

    wait_stage();
  }

  const int mbase = rowbase + wave * 16 + ((lane & 16) ? 8 : 0);
  const int n = lane & 15;
#pragma unroll
  for (int j = 0; j < 4; ++j) {
#pragma unroll
    for (int r = 0; r < 8; ++r) {
      int row = mbase + r;
      float v = acc[j][r] * comb[(long)row * CSTR + g];
      float* po = &out[(long)row * DIM + colbase + j * 16 + n];
      if (g == 0) *po = v;      // first group initializes (overwrites poison)
      else        *po += v;     // later groups accumulate (stream-serialized)
    }
  }
}

extern "C" void kernel_launch(void* const* d_in, const int* in_sizes, int n_in,
                              void* d_out, int out_size, void* d_ws, size_t ws_size,
                              hipStream_t stream) {
  const float* x     = (const float*)d_in[0];  // [B,T,D] = [N,DIM]
  const float* rw    = (const float*)d_in[1];  // [E,D]
  const float* sw1   = (const float*)d_in[2];  // [1,D,H]
  const float* sw2   = (const float*)d_in[3];  // [1,D,H]
  const float* sproj = (const float*)d_in[4];  // [1,H,D]
  const float* ew1   = (const float*)d_in[5];  // [E,D,H]
  const float* ew2   = (const float*)d_in[6];  // [E,D,H]
  const float* eproj = (const float*)d_in[7];  // [E,H,D]
  // d_in[8] = top_k (==2, baked into the router kernel)

  float* out   = (float*)d_out;                 // [N, DIM]
  float* probs = out + (size_t)N_TOK * DIM;     // [N, E]

  // workspace layout (bf16 copies + activations + combine), ~83 MB total
  char* w = (char*)d_ws;
  bf16_t* xb   = (bf16_t*)w;  w += (size_t)N_TOK * DIM * 2;
  bf16_t* w1b  = (bf16_t*)w;  w += (size_t)NGRP * DIM * HID * 2;
  bf16_t* w2b  = (bf16_t*)w;  w += (size_t)NGRP * DIM * HID * 2;
  bf16_t* wpb  = (bf16_t*)w;  w += (size_t)NGRP * HID * DIM * 2;
  bf16_t* yb   = (bf16_t*)w;  w += (size_t)N_TOK * HID * 2;
  float*  comb = (float*)w;   w += (size_t)N_TOK * CSTR * 4;

  auto cvt = [&](const float* src, bf16_t* dst, long n) {
    int blocks = (int)((n + 1023) / 1024);
    if (blocks > 4096) blocks = 4096;
    MoE_cvt<<<blocks, 256, 0, stream>>>(src, dst, n);
  };
  const long WSZ = (long)DIM * HID;
  cvt(x,     xb,        (long)N_TOK * DIM);
  cvt(sw1,   w1b,       WSZ);
  cvt(ew1,   w1b + WSZ, (long)NEXP * WSZ);
  cvt(sw2,   w2b,       WSZ);
  cvt(ew2,   w2b + WSZ, (long)NEXP * WSZ);
  cvt(sproj, wpb,       WSZ);
  cvt(eproj, wpb + WSZ, (long)NEXP * WSZ);

  MoE_router<<<N_TOK / 256, 256, 0, stream>>>(x, rw, probs, comb);

  for (int g = 0; g < NGRP; ++g) {
    const bf16_t* W1 = w1b + (size_t)g * WSZ;
    const bf16_t* W2 = w2b + (size_t)g * WSZ;
    const bf16_t* Pp = wpb + (size_t)g * WSZ;
    MoE_ffn1<<<dim3(N_TOK / BM, HID / BN), TPB, 0, stream>>>(xb, W1, W2, yb);
    MoE_ffn2<<<dim3(N_TOK / BM, DIM / BN), TPB, 0, stream>>>(yb, Pp, comb, out, g);
  }
}
